// ArcticDecoderLayer_44650480009647
// MI455X (gfx1250) — matrix operand order repaired and checked by reading, not compiled, not run on hardware
//
#include <hip/hip_runtime.h>
#include <hip/hip_bf16.h>
#include <stdint.h>

// ---------------------------------------------------------------------------
// Problem constants (ArcticDecoderLayer)
// ---------------------------------------------------------------------------
#define T_ 2048
#define H_ 2048
#define NH_ 16
#define NKV_ 4
#define HD_ 128
#define I_ 1024
#define E_ 16
#define QKV_N (NH_*HD_ + 2*NKV_*HD_)   // 3072
#define EPS_ 1e-5f

// ---------------------------------------------------------------------------
// CDNA5 async global->LDS copy (ASYNCcnt path), with portable fallback
// ---------------------------------------------------------------------------
#if defined(__AMDGCN__) && __has_builtin(__builtin_amdgcn_global_load_async_to_lds_b128)
#define HAVE_ASYNC_LDS 1
#else
#define HAVE_ASYNC_LDS 0
#endif

// CDNA5 Tensor Data Mover (TENSORcnt path)
#if defined(__AMDGCN__) && __has_builtin(__builtin_amdgcn_tensor_load_to_lds)
#define HAVE_TDM 1
#else
#define HAVE_TDM 0
#endif
#if __has_include(<hip/amd_detail/amd_gfx1250_TDM.h>)
#define TDM_SIX_ARGS 1
#else
#define TDM_SIX_ARGS 0
#endif

typedef int v4i_vs __attribute__((vector_size(16)));
typedef unsigned int u32x4 __attribute__((ext_vector_type(4)));
typedef int          i32x4 __attribute__((ext_vector_type(4)));
typedef int          i32x8 __attribute__((ext_vector_type(8)));

static __device__ __forceinline__ void async_copy16(const void* gsrc, void* ldst) {
#if HAVE_ASYNC_LDS
    __builtin_amdgcn_global_load_async_to_lds_b128(
        (__attribute__((address_space(1))) v4i_vs*)(void*)gsrc,
        (__attribute__((address_space(3))) v4i_vs*)ldst,
        0, 0);
#else
    *(uint4*)ldst = *(const uint4*)gsrc;
#endif
}
static __device__ __forceinline__ void wait_async() {
#if HAVE_ASYNC_LDS
#if __has_builtin(__builtin_amdgcn_s_wait_asynccnt)
    __builtin_amdgcn_s_wait_asynccnt(0);
#else
    asm volatile("s_wait_asynccnt 0x0" ::: "memory");
#endif
#endif
}
static __device__ __forceinline__ void wait_tensor() {
#if HAVE_TDM
#if __has_builtin(__builtin_amdgcn_s_wait_tensorcnt)
    __builtin_amdgcn_s_wait_tensorcnt(0);
#else
    asm volatile("s_wait_tensorcnt 0x0" ::: "memory");
#endif
#endif
}

// TDM 2D tile load: tile_d0 x tile_d1 elements (data_size=2B) from a row-major
// tensor with row stride stride0 (elements) into contiguous LDS at lds_addr.
// D# layout per CDNA5 ISA (groups 0/1; 2D -> groups
// 2/3 zero-filled).
static __device__ __forceinline__ void tdm_load_2d_bf16(unsigned lds_addr,
                                                        const void* gaddr,
                                                        unsigned tile_d0, unsigned tile_d1,
                                                        unsigned tensor_d0, unsigned tensor_d1,
                                                        unsigned stride0) {
#if HAVE_TDM
    const unsigned long long ga = (unsigned long long)gaddr;
    u32x4 g0;
    g0.x = 1u;                                    // count=1, user mode, no gather
    g0.y = lds_addr;                              // LDS byte address
    g0.z = (unsigned)(ga & 0xFFFFFFFFu);          // global_addr[31:0]
    g0.w = (unsigned)((ga >> 32) & 0x01FFFFFFu)   // global_addr[56:32]
         | (2u << 30);                            // type = 2 ("image")
    i32x8 g1;
    g1[0] = (int)(1u << 16);                      // workgroup_mask=0, data_size=1 (2B)
    g1[1] = (int)((tensor_d0 & 0xFFFFu) << 16);   // tensor_dim0[15:0]
    g1[2] = (int)((tensor_d0 >> 16) | ((tensor_d1 & 0xFFFFu) << 16));
    g1[3] = (int)((tensor_d1 >> 16) | (tile_d0 << 16));   // tile_dim0
    g1[4] = (int)(tile_d1 & 0xFFFFu);             // tile_dim1, tile_dim2=0
    g1[5] = (int)stride0;                         // tensor_dim0_stride[31:0]
    g1[6] = 0;                                    // stride0[47:32]=0, stride1=0
    g1[7] = 0;
    i32x4 gz4 = {0, 0, 0, 0};
#if TDM_SIX_ARGS
    i32x8 gz8 = {0, 0, 0, 0, 0, 0, 0, 0};
    __builtin_amdgcn_tensor_load_to_lds(g0, g1, gz4, gz4, gz8, 0);
#else
    __builtin_amdgcn_tensor_load_to_lds(g0, g1, gz4, gz4, 0);
#endif
#else
    (void)lds_addr; (void)gaddr; (void)tile_d0; (void)tile_d1;
    (void)tensor_d0; (void)tensor_d1; (void)stride0;
#endif
}

// ---------------------------------------------------------------------------
// Types for WMMA fragments (CDNA5 wave32, v_wmma_f32_16x16x32_bf16)
// ---------------------------------------------------------------------------
typedef __attribute__((ext_vector_type(16))) __bf16 bf16x16;
typedef __attribute__((ext_vector_type(8)))  float  f32x8;

union Frag {
    bf16x16  v;
    unsigned u[8];
};

static __device__ __forceinline__ unsigned short f2bf(float f) {
    unsigned u = __float_as_uint(f);
    unsigned r = u + 0x7FFFu + ((u >> 16) & 1u);   // round-to-nearest-even
    return (unsigned short)(r >> 16);
}
static __device__ __forceinline__ float bf2f(unsigned short h) {
    return __uint_as_float(((unsigned)h) << 16);
}
static __device__ __forceinline__ unsigned pack2(float a, float b) {
    return (unsigned)f2bf(a) | ((unsigned)f2bf(b) << 16);
}

static __device__ __forceinline__ f32x8 wmma_bf16f32(const Frag& a, const Frag& b, f32x8 c) {
    // 8 args: (neg_a, A, neg_b, B, c_mod, C, reuse_a, reuse_b)
    return __builtin_amdgcn_wmma_f32_16x16x32_bf16(false, a.v, false, b.v,
                                                   (short)0, c, false, false);
}

// ---------------------------------------------------------------------------
// Generic tiled WMMA GEMM:  C[M,N] (f32) = A[M,K] (bf16) * B (f32 weights)
//   BT == false : B stored [K,N] row-major (x @ W)
//   BT == true  : B stored [N,K] row-major (x @ W.T)
// Optional epilogue: C[row,col] += rowScale[row*rsStride] * acc   (MoE accum)
// Tile: BM=128, BN=128, BK=32; 256 threads = 8 waves (2M x 4N).
// Double-buffered LDS; A tiles staged by the Tensor Data Mover (one
// TENSOR_LOAD_TO_LDS per block per K-step, issued by wave 0), B tiles
// converted fp32->bf16 while staging. Next tile in flight during WMMAs.
// Requires M%128==0, N%128==0, K%32==0 (true for every use in this layer).
// ---------------------------------------------------------------------------
template <bool BT>
__global__ __launch_bounds__(256)
void gemm_bf16(const unsigned short* __restrict__ A,
               const float* __restrict__ B,
               float* __restrict__ C,
               int M, int N, int K,
               const float* __restrict__ rowScale, int rsStride)
{
    __shared__ unsigned short As[2][128][32];  // [m][k], pairs contiguous in k
    __shared__ unsigned short Bs[2][128][32];  // [n][k], pairs contiguous in k

    const int tid  = threadIdx.x;
    const int lane = tid & 31;
    const int wave = tid >> 5;
    const int g    = lane >> 4;      // lane half
    const int ln   = lane & 15;
    const int m0   = blockIdx.y * 128;
    const int n0   = blockIdx.x * 128;
    const int wm   = (wave & 1) * 64;    // 2 waves across M
    const int wn   = (wave >> 1) * 32;   // 4 waves across N

    auto stageA = [&](int k0, int buf) {
#if HAVE_TDM
        if (wave == 0) {   // TDM ignores EXEC; one DMA op per block
            const unsigned lds = (unsigned)(unsigned long long)
                (__attribute__((address_space(3))) void*)&As[buf][0][0];
            tdm_load_2d_bf16(lds, A + (size_t)m0 * K + k0,
                             /*tile*/32, 128,
                             /*tensor*/(unsigned)K, (unsigned)M,
                             /*row stride*/(unsigned)K);
        }
#else
        const int row = tid >> 1, seg = tid & 1;
        const unsigned short* src = A + (size_t)(m0 + row) * K + k0 + seg * 16;
        async_copy16(src,     &As[buf][row][seg * 16]);
        async_copy16(src + 8, &As[buf][row][seg * 16 + 8]);
#endif
    };
    auto waitA = [&]() {
#if HAVE_TDM
        if (wave == 0) wait_tensor();
#else
        wait_async();
#endif
    };
    auto stageB = [&](int k0, int buf) {
        if (BT) {
            const int n = tid >> 1, seg = tid & 1;
            const float* src = B + (size_t)(n0 + n) * K + k0 + seg * 16;
            union { float4 q[4]; float f[16]; } t;
            t.q[0] = ((const float4*)src)[0];
            t.q[1] = ((const float4*)src)[1];
            t.q[2] = ((const float4*)src)[2];
            t.q[3] = ((const float4*)src)[3];
            unsigned* dst = (unsigned*)&Bs[buf][n][seg * 16];
#pragma unroll
            for (int j = 0; j < 8; ++j) dst[j] = pack2(t.f[2 * j], t.f[2 * j + 1]);
        } else {
            const int k = tid >> 3, nsg = tid & 7;
            const float* src = B + (size_t)(k0 + k) * N + n0 + nsg * 16;
            union { float4 q[4]; float f[16]; } t;
            t.q[0] = ((const float4*)src)[0];
            t.q[1] = ((const float4*)src)[1];
            t.q[2] = ((const float4*)src)[2];
            t.q[3] = ((const float4*)src)[3];
#pragma unroll
            for (int i = 0; i < 16; ++i) Bs[buf][nsg * 16 + i][k] = f2bf(t.f[i]);
        }
    };

    f32x8 acc[4][2];
#pragma unroll
    for (int ms = 0; ms < 4; ++ms)
#pragma unroll
        for (int ns = 0; ns < 2; ++ns)
#pragma unroll
            for (int r = 0; r < 8; ++r) acc[ms][ns][r] = 0.0f;

    const int nkt = K >> 5;
    stageA(0, 0);
    stageB(0, 0);

    for (int kt = 0; kt < nkt; ++kt) {
        const int k0  = kt << 5;
        const int buf = kt & 1;

        waitA();               // A tile for As[buf] landed (TDM / async)
        __syncthreads();       // all waves' As/Bs[buf] writes visible

        // stage next tile into the other buffer while computing this one
        if (kt + 1 < nkt) {
            stageA(k0 + 32, buf ^ 1);
            stageB(k0 + 32, buf ^ 1);
        }
        // prefetch two tiles ahead (global_prefetch_b8)
        if (kt + 2 < nkt) {
            __builtin_prefetch(A + (size_t)(m0 + (tid >> 1)) * K + k0 + 64, 0, 1);
            if (BT)
                __builtin_prefetch(B + (size_t)(n0 + (tid >> 1)) * K + k0 + 64, 0, 1);
            else
                __builtin_prefetch(B + (size_t)(k0 + 64 + (tid >> 3)) * N + n0, 0, 1);
        }

        // ---- fragments (ISA 7.12.2 16-bit A / B layouts) + 8 WMMAs ----
        Frag af[4];
#pragma unroll
        for (int ms = 0; ms < 4; ++ms) {
            const int row = wm + ms * 16 + ln;
#pragma unroll
            for (int r = 0; r < 8; ++r) {
                const int kk = (r < 4) ? (2 * r + 8 * g) : (16 + 2 * (r - 4) + 8 * g);
                af[ms].u[r] = *(const unsigned*)&As[buf][row][kk];
            }
        }
        Frag bfr[2];
#pragma unroll
        for (int ns = 0; ns < 2; ++ns) {
            const int col = wn + ns * 16 + ln;
#pragma unroll
            for (int r = 0; r < 8; ++r) {
                const int kk = (r < 4) ? (2 * r + 8 * g) : (16 + 2 * (r - 4) + 8 * g);
                bfr[ns].u[r] = *(const unsigned*)&Bs[buf][col][kk];
            }
        }
#pragma unroll
        for (int ms = 0; ms < 4; ++ms)
#pragma unroll
            for (int ns = 0; ns < 2; ++ns)
                acc[ms][ns] = wmma_bf16f32(af[ms], bfr[ns], acc[ms][ns]);
    }

    // ---- epilogue (C layout: VGPR r -> M = r + 8*half, N = lane&15) ----
#pragma unroll
    for (int ms = 0; ms < 4; ++ms)
#pragma unroll
        for (int ns = 0; ns < 2; ++ns)
#pragma unroll
            for (int r = 0; r < 8; ++r) {
                const int row = m0 + wm + ms * 16 + r + 8 * g;
                const int col = n0 + wn + ns * 16 + ln;
                const size_t idx = (size_t)row * N + col;
                const float v = acc[ms][ns][r];
                if (rowScale) C[idx] += rowScale[(size_t)row * rsStride] * v;
                else          C[idx] = v;
            }
}

// ---------------------------------------------------------------------------
// RMSNorm: y_bf16 = x * rsqrt(mean(x^2)+eps) * w     (one block per token)
// ---------------------------------------------------------------------------
__global__ __launch_bounds__(256)
void rmsnorm_bf16(const float* __restrict__ X, const float* __restrict__ W,
                  unsigned short* __restrict__ Y)
{
    __shared__ float red[256];
    const int t = blockIdx.x, tid = threadIdx.x;
    const float* x = X + (size_t)t * H_;
    float ss = 0.f;
#pragma unroll
    for (int i = 0; i < H_ / 256; ++i) { float v = x[tid + i * 256]; ss += v * v; }
    red[tid] = ss; __syncthreads();
    for (int s = 128; s > 0; s >>= 1) { if (tid < s) red[tid] += red[tid + s]; __syncthreads(); }
    const float scale = rsqrtf(red[0] / (float)H_ + EPS_);
    unsigned short* y = Y + (size_t)t * H_;
#pragma unroll
    for (int i = 0; i < H_ / 256; ++i) {
        const int c = tid + i * 256;
        y[c] = f2bf(x[c] * scale * W[c]);
    }
}

// ---------------------------------------------------------------------------
// RoPE (NeoX) on Q/K from the fused QKV buffer + bf16 conversion of Q/K/V.
// One block (1024 threads) per token.
// ---------------------------------------------------------------------------
__global__ __launch_bounds__(1024)
void rope_qkv(const float* __restrict__ QKV, const long long* __restrict__ pos,
              unsigned short* __restrict__ Q, unsigned short* __restrict__ K,
              unsigned short* __restrict__ V)
{
    const int t = blockIdx.x, tid = threadIdx.x;
    const float p = (float)pos[t];
    const float* base = QKV + (size_t)t * QKV_N;
    {   // Q: 16 heads x 64 rotation pairs
        const int h = tid >> 6, i = tid & 63;
        const float inv = __powf(10000.f, -(float)i / 64.f);
        float sn, cs; __sincosf(p * inv, &sn, &cs);
        const float x1 = base[h * HD_ + i];
        const float x2 = base[h * HD_ + 64 + i];
        Q[(size_t)t * (NH_ * HD_) + h * HD_ + i]      = f2bf(x1 * cs - x2 * sn);
        Q[(size_t)t * (NH_ * HD_) + h * HD_ + 64 + i] = f2bf(x2 * cs + x1 * sn);
    }
    if (tid < NKV_ * 64) {  // K: 4 kv heads x 64 pairs
        const int h = tid >> 6, i = tid & 63;
        const float inv = __powf(10000.f, -(float)i / 64.f);
        float sn, cs; __sincosf(p * inv, &sn, &cs);
        const float* kb = base + NH_ * HD_;
        const float x1 = kb[h * HD_ + i];
        const float x2 = kb[h * HD_ + 64 + i];
        K[(size_t)t * (NKV_ * HD_) + h * HD_ + i]      = f2bf(x1 * cs - x2 * sn);
        K[(size_t)t * (NKV_ * HD_) + h * HD_ + 64 + i] = f2bf(x2 * cs + x1 * sn);
    }
    if (tid < NKV_ * HD_) { // V passthrough -> bf16
        V[(size_t)t * (NKV_ * HD_) + tid] = f2bf(base[NH_ * HD_ + NKV_ * HD_ + tid]);
    }
}

// ---------------------------------------------------------------------------
// Flash attention (causal, GQA 4:1). Grid (T/128, NH), 256 threads = 8 waves,
// each wave owns 16 query rows. 64-key tiles staged in LDS (K tile via the
// async-to-LDS path); S and O built with v_wmma_f32_16x16x32_bf16; online
// softmax exploits matching C-matrix row mapping between S and O accumulators.
// ---------------------------------------------------------------------------
__global__ __launch_bounds__(256)
void flash_attn(const unsigned short* __restrict__ Q,
                const unsigned short* __restrict__ Kb,
                const unsigned short* __restrict__ Vb,
                unsigned short* __restrict__ O, float scale)
{
    __shared__ unsigned short Kt[64][HD_];   // [key][hd]     16 KB
    __shared__ unsigned short Vt[HD_][64];   // [hd][key] (T) 16 KB
    __shared__ unsigned short Pt[8][16][64]; // per-wave P    16 KB

    const int head = blockIdx.y;
    const int qb0  = blockIdx.x * 128;
    const int tid  = threadIdx.x;
    const int lane = tid & 31, wave = tid >> 5;
    const int g = lane >> 4, ln = lane & 15;
    const int wrow0 = qb0 + wave * 16;
    const int kvh = head >> 2;               // NH/NKV = 4

    // Cache Q fragments for this wave's 16 rows (4 K-steps over HD=128)
    Frag qf[4];
    {
        const unsigned short* qp = Q + (size_t)(wrow0 + ln) * (NH_ * HD_) + head * HD_;
#pragma unroll
        for (int ks = 0; ks < 4; ++ks)
#pragma unroll
            for (int r = 0; r < 8; ++r) {
                const int kk = ks * 32 + ((r < 4) ? (2 * r + 8 * g) : (16 + 2 * (r - 4) + 8 * g));
                qf[ks].u[r] = *(const unsigned*)(qp + kk);
            }
    }

    f32x8 Oa[8];
    float mrow[8], lrow[8];
#pragma unroll
    for (int n8 = 0; n8 < 8; ++n8)
#pragma unroll
        for (int r = 0; r < 8; ++r) Oa[n8][r] = 0.0f;
#pragma unroll
    for (int r = 0; r < 8; ++r) { mrow[r] = -3.0e38f; lrow[r] = 0.0f; }

    const int nkb = (qb0 + 127) / 64 + 1;
    for (int kb = 0; kb < nkb; ++kb) {
        const int keys0 = kb * 64;

        // ---- stage K (async global->LDS) and V (register transpose) ----
        {
            const int key = tid >> 2, c = tid & 3;
            const unsigned short* ksrc =
                Kb + (size_t)(keys0 + key) * (NKV_ * HD_) + kvh * HD_ + c * 32;
#pragma unroll
            for (int j = 0; j < 4; ++j)
                async_copy16(ksrc + j * 8, &Kt[key][c * 32 + j * 8]);

            const unsigned short* vsrc =
                Vb + (size_t)(keys0 + key) * (NKV_ * HD_) + kvh * HD_ + c * 32;
            union { uint4 q[4]; unsigned short s[32]; } vv;
#pragma unroll
            for (int j = 0; j < 4; ++j) vv.q[j] = *(const uint4*)(vsrc + j * 8);
#pragma unroll
            for (int i = 0; i < 32; ++i) Vt[c * 32 + i][key] = vv.s[i];
        }
        wait_async();
        __syncthreads();

        const bool active = (keys0 <= wrow0 + 15);
        f32x8 S[4];
        if (active) {
#pragma unroll
            for (int nt = 0; nt < 4; ++nt)
#pragma unroll
                for (int r = 0; r < 8; ++r) S[nt][r] = 0.0f;

            // S = Q * K^T  (B-frag: contraction over HD, N = key)
#pragma unroll
            for (int ks = 0; ks < 4; ++ks)
#pragma unroll
                for (int nt = 0; nt < 4; ++nt) {
                    Frag kf;
#pragma unroll
                    for (int r = 0; r < 8; ++r) {
                        const int kk = ks * 32 + ((r < 4) ? (2 * r + 8 * g) : (16 + 2 * (r - 4) + 8 * g));
                        kf.u[r] = *(const unsigned*)&Kt[nt * 16 + ln][kk];
                    }
                    S[nt] = wmma_bf16f32(qf[ks], kf, S[nt]);
                }

            // scale + causal mask
#pragma unroll
            for (int nt = 0; nt < 4; ++nt)
#pragma unroll
                for (int r = 0; r < 8; ++r) {
                    const int colg = keys0 + nt * 16 + ln;
                    const int rowg = wrow0 + r + 8 * g;
                    float s = S[nt][r] * scale;
                    if (colg > rowg) s = -3.0e38f;
                    S[nt][r] = s;
                }

            // online softmax (row reductions across each 16-lane half)
            float al[8];
#pragma unroll
            for (int r = 0; r < 8; ++r) {
                float mx = fmaxf(fmaxf(S[0][r], S[1][r]), fmaxf(S[2][r], S[3][r]));
#pragma unroll
                for (int off = 8; off > 0; off >>= 1) mx = fmaxf(mx, __shfl_xor(mx, off, 32));
                const float mn = fmaxf(mrow[r], mx);
                al[r] = __expf(mrow[r] - mn);
                mrow[r] = mn;
                float sum = 0.f;
#pragma unroll
                for (int nt = 0; nt < 4; ++nt) {
                    const float p = __expf(S[nt][r] - mn);
                    S[nt][r] = p; sum += p;
                }
#pragma unroll
                for (int off = 8; off > 0; off >>= 1) sum += __shfl_xor(sum, off, 32);
                lrow[r] = lrow[r] * al[r] + sum;
            }
#pragma unroll
            for (int n8 = 0; n8 < 8; ++n8)
#pragma unroll
                for (int r = 0; r < 8; ++r) Oa[n8][r] *= al[r];

            // P -> LDS (C layout -> A layout reshuffle)
#pragma unroll
            for (int nt = 0; nt < 4; ++nt)
#pragma unroll
                for (int r = 0; r < 8; ++r)
                    Pt[wave][r + 8 * g][nt * 16 + ln] = f2bf(S[nt][r]);
        }
        __syncthreads();

        if (active) {
            // O += P * V  (contraction over 64 keys = 2 K-steps)
#pragma unroll
            for (int ks2 = 0; ks2 < 2; ++ks2) {
                Frag pf;
#pragma unroll
                for (int r = 0; r < 8; ++r) {
                    const int kk = ks2 * 32 + ((r < 4) ? (2 * r + 8 * g)
                                                       : (16 + 2 * (r - 4) + 8 * g));
                    pf.u[r] = *(const unsigned*)&Pt[wave][ln][kk];
                }
#pragma unroll
                for (int n8 = 0; n8 < 8; ++n8) {
                    Frag vf;
#pragma unroll
                    for (int r = 0; r < 8; ++r) {
                        const int kk = ks2 * 32 + ((r < 4) ? (2 * r + 8 * g) : (16 + 2 * (r - 4) + 8 * g));
                        vf.u[r] = *(const unsigned*)&Vt[n8 * 16 + ln][kk];
                    }
                    Oa[n8] = wmma_bf16f32(pf, vf, Oa[n8]);
                }
            }
        }
        __syncthreads();
    }

    // epilogue: O /= l, store bf16 into [T, NH*HD]
#pragma unroll
    for (int n8 = 0; n8 < 8; ++n8)
#pragma unroll
        for (int r = 0; r < 8; ++r) {
            const int rowg = wrow0 + r + 8 * g;
            const int colg = head * HD_ + n8 * 16 + ln;
            O[(size_t)rowg * (NH_ * HD_) + colg] = f2bf(Oa[n8][r] / lrow[r]);
        }
}

// ---------------------------------------------------------------------------
// MoE gate: logits = x @ gate_w, softmax, top-2, renormalize.
// Writes dense per-token weights W[t][e] (zeros for unselected experts).
// ---------------------------------------------------------------------------
__global__ __launch_bounds__(256)
void gate_topk(const unsigned short* __restrict__ X, const float* __restrict__ GW,
               float* __restrict__ Wout)
{
    __shared__ float sm[256];
    __shared__ float logits[E_];
    const int t = blockIdx.x, tid = threadIdx.x;
    const int e = tid & 15, j = tid >> 4;
    float acc = 0.f;
    const unsigned short* x = X + (size_t)t * H_;
#pragma unroll 4
    for (int k = 0; k < H_ / 16; ++k) {
        const int c = j * (H_ / 16) + k;
        acc += bf2f(x[c]) * GW[(size_t)c * E_ + e];
    }
    sm[tid] = acc; __syncthreads();
    if (tid < E_) {
        float s = 0.f;
#pragma unroll
        for (int jj = 0; jj < 16; ++jj) s += sm[jj * 16 + tid];
        logits[tid] = s;
    }
    __syncthreads();
    if (tid == 0) {
        float lmax = -3.0e38f;
        for (int k = 0; k < E_; ++k) lmax = fmaxf(lmax, logits[k]);
        float p[E_]; float psum = 0.f;
        for (int k = 0; k < E_; ++k) { p[k] = __expf(logits[k] - lmax); psum += p[k]; }
        for (int k = 0; k < E_; ++k) p[k] /= psum;
        int i1 = 0; for (int k = 1; k < E_; ++k) if (p[k] > p[i1]) i1 = k;
        int i2 = (i1 == 0) ? 1 : 0;
        for (int k = 0; k < E_; ++k) if (k != i1 && p[k] > p[i2]) i2 = k;
        const float wsum = p[i1] + p[i2];
        for (int k = 0; k < E_; ++k)
            Wout[(size_t)t * E_ + k] = (k == i1) ? p[i1] / wsum
                                    : (k == i2) ? p[i2] / wsum : 0.f;
    }
}

// ---------------------------------------------------------------------------
// Elementwise kernels
// ---------------------------------------------------------------------------
__global__ void silu_mul_k(const float* __restrict__ GU, unsigned short* __restrict__ Hout,
                           int X, int total)
{
    int idx = blockIdx.x * blockDim.x + threadIdx.x;
    if (idx >= total) return;
    const int t = idx / X, c = idx - t * X;
    const float gv = GU[(size_t)t * 2 * X + c];
    const float uv = GU[(size_t)t * 2 * X + X + c];
    const float s = gv / (1.f + __expf(-gv));
    Hout[idx] = f2bf(s * uv);
}
__global__ void add2_k(const float* __restrict__ A, const float* __restrict__ B,
                       float* __restrict__ C, int n)
{
    int i = blockIdx.x * blockDim.x + threadIdx.x;
    if (i < n) C[i] = A[i] + B[i];
}
__global__ void add3_k(const float* __restrict__ A, const float* __restrict__ B,
                       const float* __restrict__ C, float* __restrict__ D, int n)
{
    int i = blockIdx.x * blockDim.x + threadIdx.x;
    if (i < n) D[i] = A[i] + B[i] + C[i];
}
__global__ void zero_k(float* __restrict__ A, int n)
{
    int i = blockIdx.x * blockDim.x + threadIdx.x;
    if (i < n) A[i] = 0.f;
}

// ---------------------------------------------------------------------------
// Host orchestration
// ---------------------------------------------------------------------------
extern "C" void kernel_launch(void* const* d_in, const int* in_sizes, int n_in,
                              void* d_out, int out_size, void* d_ws, size_t ws_size,
                              hipStream_t stream)
{
    (void)in_sizes; (void)n_in; (void)out_size; (void)ws_size;

    const float* hidden  = (const float*)d_in[0];
    const float* ln_in   = (const float*)d_in[1];
    const float* ln_post = (const float*)d_in[2];
    const float* ln_res  = (const float*)d_in[3];
    const float* wqkv    = (const float*)d_in[4];
    const float* wo      = (const float*)d_in[5];
    const float* res_w13 = (const float*)d_in[6];
    const float* res_w2  = (const float*)d_in[7];
    const float* gate_w  = (const float*)d_in[8];
    const float* ws      = (const float*)d_in[9];
    const float* w2s     = (const float*)d_in[10];
    const long long* pos = (const long long*)d_in[11];
    float* out = (float*)d_out;

    // Scratch carve-up (256B aligned)
    char* base = (char*)d_ws;
    size_t off = 0;
    auto alloc = [&](size_t bytes) -> char* {
        off = (off + 255) & ~(size_t)255;
        char* p = base + off;
        off += bytes;
        return p;
    };
    unsigned short* x_ln    = (unsigned short*)alloc((size_t)T_ * H_ * 2);
    float*          qkv     = (float*)         alloc((size_t)T_ * QKV_N * 4);
    unsigned short* q_bf    = (unsigned short*)alloc((size_t)T_ * NH_ * HD_ * 2);
    unsigned short* k_bf    = (unsigned short*)alloc((size_t)T_ * NKV_ * HD_ * 2);
    unsigned short* v_bf    = (unsigned short*)alloc((size_t)T_ * NKV_ * HD_ * 2);
    unsigned short* attn_o  = (unsigned short*)alloc((size_t)T_ * H_ * 2);
    float*          attn_pj = (float*)         alloc((size_t)T_ * H_ * 4);
    float*          res_at  = (float*)         alloc((size_t)T_ * H_ * 4);
    unsigned short* h2_bf   = (unsigned short*)alloc((size_t)T_ * H_ * 2);
    float*          gu      = (float*)         alloc((size_t)T_ * 2 * H_ * 4);
    unsigned short* hmid    = (unsigned short*)alloc((size_t)T_ * H_ * 2);
    float*          resmlp  = (float*)         alloc((size_t)T_ * H_ * 4);
    unsigned short* xpost   = (unsigned short*)alloc((size_t)T_ * H_ * 2);
    float*          gatew   = (float*)         alloc((size_t)T_ * E_ * 4);
    float*          gu_e    = (float*)         alloc((size_t)T_ * 2 * I_ * 4);
    unsigned short* he_bf   = (unsigned short*)alloc((size_t)T_ * I_ * 2);
    float*          moe_out = (float*)         alloc((size_t)T_ * H_ * 4);

    auto gemm = [&](bool bt, const unsigned short* A, const float* B, float* C,
                    int M, int N, int K, const float* rs, int rss) {
        dim3 grid(N / 128, M / 128);
        if (bt) gemm_bf16<true ><<<grid, 256, 0, stream>>>(A, B, C, M, N, K, rs, rss);
        else    gemm_bf16<false><<<grid, 256, 0, stream>>>(A, B, C, M, N, K, rs, rss);
    };
    const int nTH = T_ * H_;

    // ---- attention branch ----
    rmsnorm_bf16<<<T_, 256, 0, stream>>>(hidden, ln_in, x_ln);
    gemm(false, x_ln, wqkv, qkv, T_, QKV_N, H_, nullptr, 0);
    rope_qkv<<<T_, 1024, 0, stream>>>(qkv, pos, q_bf, k_bf, v_bf);
    {
        dim3 grid(T_ / 128, NH_);
        flash_attn<<<grid, 256, 0, stream>>>(q_bf, k_bf, v_bf, attn_o, 0.08838834764831845f);
    }
    gemm(false, attn_o, wo, attn_pj, T_, H_, NH_ * HD_, nullptr, 0);
    add2_k<<<(nTH + 255) / 256, 256, 0, stream>>>(hidden, attn_pj, res_at, nTH);

    // ---- residual MLP branch ----
    rmsnorm_bf16<<<T_, 256, 0, stream>>>(res_at, ln_res, h2_bf);
    gemm(false, h2_bf, res_w13, gu, T_, 2 * H_, H_, nullptr, 0);
    silu_mul_k<<<(nTH + 255) / 256, 256, 0, stream>>>(gu, hmid, H_, nTH);
    gemm(false, hmid, res_w2, resmlp, T_, H_, H_, nullptr, 0);

    // ---- MoE branch ----
    rmsnorm_bf16<<<T_, 256, 0, stream>>>(hidden, ln_post, xpost);
    gate_topk<<<T_, 256, 0, stream>>>(xpost, gate_w, gatew);
    zero_k<<<(nTH + 255) / 256, 256, 0, stream>>>(moe_out, nTH);
    const int nTI = T_ * I_;
    for (int e = 0; e < E_; ++e) {
        const float* wse  = ws  + (size_t)e * 2 * I_ * H_;  // [2I, H]
        const float* w2se = w2s + (size_t)e * H_ * I_;      // [H, I]
        gemm(true, xpost, wse, gu_e, T_, 2 * I_, H_, nullptr, 0);
        silu_mul_k<<<(nTI + 255) / 256, 256, 0, stream>>>(gu_e, he_bf, I_, nTI);
        gemm(true, he_bf, w2se, moe_out, T_, H_, I_, gatew + e, E_);
    }

    // ---- final combine: out = residual_attn + residual_mlp + moe ----
    add3_k<<<(nTH + 255) / 256, 256, 0, stream>>>(res_at, resmlp, moe_out, out, nTH);
}
